// ScatterCFGEncodedPathsToCFGNodeEncodings_41987600285775
// MI455X (gfx1250) — compile-verified, matrix operand order checked by reading, not run on hardware
//
#include <hip/hip_runtime.h>
#include <hip/hip_bf16.h>

// ---------------------------------------------------------------------------
// Problem constants (match reference)
// ---------------------------------------------------------------------------
constexpr int   D    = 256;       // node encoding dim
constexpr int   NP   = 2048;      // n paths
constexpr int   PL   = 256;       // path len
constexpr int   NN   = 50000;     // n cfg nodes
constexpr float NEG_BIG = -1e30f;
constexpr size_t NE = (size_t)NP * PL;          // 524288 occurrences
constexpr int   LPAD = 260;                     // LDS row pitch (f32) — bank-conflict pad

typedef __bf16 bf16_t;
typedef __attribute__((ext_vector_type(16))) __bf16 v16bf;
typedef __attribute__((ext_vector_type(8)))  float  v8f;
typedef __attribute__((ext_vector_type(4)))  float  v4f;
typedef __attribute__((ext_vector_type(4)))  int    v4i;

#define AS1 __attribute__((address_space(1)))
#define AS3 __attribute__((address_space(3)))

#if __has_builtin(__builtin_amdgcn_global_load_async_to_lds_b128) && \
    __has_builtin(__builtin_amdgcn_s_wait_asynccnt)
#define USE_ASYNC_LDS 1
#else
#define USE_ASYNC_LDS 0
#endif

// float -> bf16 via native convert (backend emits v_cvt_*bf16_f32)
__device__ __forceinline__ bf16_t f2bf(float f) { return (bf16_t)f; }

// order-preserving float<->uint mapping for atomicMax on floats
__device__ __forceinline__ unsigned flipf(float f) {
    unsigned b = __float_as_uint(f);
    return (b & 0x80000000u) ? ~b : (b | 0x80000000u);
}
__device__ __forceinline__ float unflipf(unsigned u) {
    unsigned b = (u & 0x80000000u) ? (u ^ 0x80000000u) : ~u;
    return __uint_as_float(b);
}

// fragment K offset pattern for 16-bit A/B operands of wmma_*_16x16x32
__device__ __forceinline__ int frag_k0(int v, int half) {
    return half * 8 + (v & 3) * 2 + ((v >> 2) * 16);
}

// stage one 16-row x 256-col f32 tile (contiguous rows) into padded LDS.
// One wave: 512B per async instruction, fully coalesced.
__device__ __forceinline__ void stage_tile(const float* __restrict__ gsrc,
                                           float* __restrict__ lds, int lane) {
#if USE_ASYNC_LDS
#pragma unroll
    for (int r = 0; r < 16; ++r) {
        const char* g = (const char*)(gsrc + (size_t)r * D);
        char*       l = (char*)(lds + r * LPAD);
        __builtin_amdgcn_global_load_async_to_lds_b128(
            (AS1 v4i*)(g + lane * 16), (AS3 v4i*)(l + lane * 16), 0, 0);
        __builtin_amdgcn_global_load_async_to_lds_b128(
            (AS1 v4i*)(g + 512 + lane * 16), (AS3 v4i*)(l + 512 + lane * 16), 0, 0);
    }
#else
#pragma unroll
    for (int r = 0; r < 16; ++r) {
        v4f a = *(const v4f*)(gsrc + (size_t)r * D + lane * 4);
        v4f b = *(const v4f*)(gsrc + (size_t)r * D + 128 + lane * 4);
        *(v4f*)(lds + r * LPAD + lane * 4)       = a;
        *(v4f*)(lds + r * LPAD + 128 + lane * 4) = b;
    }
#endif
}
__device__ __forceinline__ void stage_wait() {
#if USE_ASYNC_LDS
    __builtin_amdgcn_s_wait_asynccnt(0);
#endif
}

// build 16x32 bf16 A fragment from padded LDS tile
__device__ __forceinline__ v16bf make_afrag(const float* __restrict__ lds,
                                            int ks, int lane) {
    int half = lane >> 4;
    const float* row = lds + (lane & 15) * LPAD + ks * 32;
    v16bf a;
#pragma unroll
    for (int v = 0; v < 8; ++v) {
        int k0 = frag_k0(v, half);
        float2 f = *(const float2*)(row + k0);
        a[2 * v]     = f2bf(f.x);
        a[2 * v + 1] = f2bf(f.y);
    }
    return a;
}

// ---------------------------------------------------------------------------
// 0) init: zero agg accumulator (d_out), init segmax/denom
// ---------------------------------------------------------------------------
__global__ __launch_bounds__(256) void k_init(float* __restrict__ agg,
                                              unsigned* __restrict__ segmax,
                                              float* __restrict__ denom) {
    size_t i = (size_t)blockIdx.x * 256 + threadIdx.x;   // 50000*256 threads
    agg[i] = 0.0f;
    if (i < (size_t)NN) {
        segmax[i] = flipf(NEG_BIG);
        denom[i]  = 0.0f;
    }
}

// ---------------------------------------------------------------------------
// 1) pack a [K,256] f32 weight matrix into WMMA B-fragment order (bf16)
// ---------------------------------------------------------------------------
__global__ __launch_bounds__(256) void k_pack(const float* __restrict__ W,
                                              int nfrag,
                                              bf16_t* __restrict__ out) {
    int frag = blockIdx.x * 8 + (threadIdx.x >> 5);
    if (frag >= nfrag) return;
    int ln   = threadIdx.x & 31;
    int ks   = frag >> 4;
    int nt   = frag & 15;
    int half = ln >> 4;
    int n    = nt * 16 + (ln & 15);
    bf16_t* dst = out + ((size_t)frag * 32 + ln) * 16;
#pragma unroll
    for (int v = 0; v < 8; ++v) {
        int k0 = ks * 32 + frag_k0(v, half);
        dst[2 * v]     = f2bf(W[(size_t)k0 * D + n]);
        dst[2 * v + 1] = f2bf(W[(size_t)(k0 + 1) * D + n]);
    }
}

// ---------------------------------------------------------------------------
// 2) q = prev @ Wq  (bf16 WMMA, f32 acc); one wave = 16 rows, A staged in LDS
// ---------------------------------------------------------------------------
__global__ __launch_bounds__(32) void k_qproj(const float* __restrict__ prev,
                                              const bf16_t* __restrict__ WqP,
                                              float* __restrict__ q) {
    __shared__ float s_a[16 * LPAD];
    int m0 = blockIdx.x * 16;
    int ln = threadIdx.x;

    stage_tile(prev + (size_t)m0 * D, s_a, ln);
    stage_wait();

    const v16bf* B = (const v16bf*)WqP;   // fragment f: 32 lanes x 1 v16bf
    v8f acc[16];
#pragma unroll
    for (int t = 0; t < 16; ++t) acc[t] = (v8f){0,0,0,0,0,0,0,0};

    for (int ks = 0; ks < 8; ++ks) {
        v16bf afrag = make_afrag(s_a, ks, ln);
        v16bf bcur = B[((size_t)(ks * 16)) * 32 + ln];
#pragma unroll
        for (int nt = 0; nt < 16; ++nt) {
            v16bf bnxt = bcur;
            if (nt < 15) bnxt = B[((size_t)(ks * 16 + nt + 1)) * 32 + ln];
            acc[nt] = __builtin_amdgcn_wmma_f32_16x16x32_bf16(
                false, afrag, false, bcur, (short)0, acc[nt], false, false);
            bcur = bnxt;
        }
    }
    int col = ln & 15, half = ln >> 4;
#pragma unroll
    for (int nt = 0; nt < 16; ++nt)
#pragma unroll
        for (int r = 0; r < 8; ++r)
            q[(size_t)(m0 + half * 8 + r) * D + nt * 16 + col] = acc[nt][r];
}

// ---------------------------------------------------------------------------
// 3) scores: one wave per occurrence; dot(enc[e], q[idx[e]])/16, mask, segmax
//    enc streamed non-temporal (537MB, read-once; keep q/agg resident in L2)
// ---------------------------------------------------------------------------
__global__ __launch_bounds__(256) void k_scores(const float* __restrict__ enc,
                                                const unsigned char* __restrict__ mask,
                                                const int* __restrict__ idx,
                                                const float* __restrict__ q,
                                                float* __restrict__ scores,
                                                unsigned* __restrict__ segmax) {
    size_t e  = (size_t)blockIdx.x * 8 + (threadIdx.x >> 5);
    int lane  = threadIdx.x & 31;
    int node  = idx[e];
    const float* er = enc + e * D + lane * 8;
    const float* qr = q + (size_t)node * D + lane * 8;
    v4f a0 = __builtin_nontemporal_load((const v4f*)er);
    v4f a1 = __builtin_nontemporal_load((const v4f*)(er + 4));
    v4f b0 = *(const v4f*)qr;
    v4f b1 = *(const v4f*)(qr + 4);
    float s = a0.x * b0.x + a0.y * b0.y + a0.z * b0.z + a0.w * b0.w
            + a1.x * b1.x + a1.y * b1.y + a1.z * b1.z + a1.w * b1.w;
#pragma unroll
    for (int off = 16; off > 0; off >>= 1) s += __shfl_xor(s, off, 32);
    if (lane == 0) {
        s = mask[e] ? (s * 0.0625f) : NEG_BIG;   // 1/sqrt(256) = 1/16
        scores[e] = s;
        atomicMax(&segmax[node], flipf(s));
    }
}

// ---------------------------------------------------------------------------
// 4) ex = exp(s - segmax'), denom += ex
// ---------------------------------------------------------------------------
__global__ __launch_bounds__(256) void k_expden(const int* __restrict__ idx,
                                                const unsigned* __restrict__ segmax,
                                                float* __restrict__ scores,
                                                float* __restrict__ denom) {
    size_t e = (size_t)blockIdx.x * 256 + threadIdx.x;
    int node = idx[e];
    float m = unflipf(segmax[node]);
    m = (m > NEG_BIG * 0.5f) ? m : 0.0f;          // guard empty segments
    float ex = __expf(scores[e] - m);             // masked -> exp(-huge) = 0
    scores[e] = ex;
    atomicAdd(&denom[node], ex);
}

// ---------------------------------------------------------------------------
// 5) agg[idx] += (ex/denom) * enc   (scatter-add into d_out)
// ---------------------------------------------------------------------------
__global__ __launch_bounds__(256) void k_agg(const float* __restrict__ enc,
                                             const int* __restrict__ idx,
                                             const float* __restrict__ scores,
                                             const float* __restrict__ denom,
                                             float* __restrict__ agg) {
    size_t e = (size_t)blockIdx.x * 8 + (threadIdx.x >> 5);
    int lane = threadIdx.x & 31;
    int node = idx[e];
    float w = scores[e] / (denom[node] + 1e-9f);
    const float* er = enc + e * D + lane * 8;
    float*       ar = agg + (size_t)node * D + lane * 8;
    v4f a0 = __builtin_nontemporal_load((const v4f*)er);
    v4f a1 = __builtin_nontemporal_load((const v4f*)(er + 4));
    float vals[8] = {a0.x, a0.y, a0.z, a0.w, a1.x, a1.y, a1.z, a1.w};
#pragma unroll
    for (int j = 0; j < 8; ++j) atomicAdd(ar + j, w * vals[j]);
}

// ---------------------------------------------------------------------------
// 6) gate: z = sigmoid([prev;agg] @ Wg + bg); out = z*prev + (1-z)*agg
//    agg lives in d_out; both 16-row tiles staged in LDS; in-place update.
// ---------------------------------------------------------------------------
__global__ __launch_bounds__(32) void k_gate(const float* __restrict__ prev,
                                             const bf16_t* __restrict__ WgP,
                                             const float* __restrict__ bg,
                                             float* __restrict__ out) {
    __shared__ float s_p[16 * LPAD];   // prev tile
    __shared__ float s_g[16 * LPAD];   // agg tile
    int m0 = blockIdx.x * 16;
    int ln = threadIdx.x;

    stage_tile(prev + (size_t)m0 * D, s_p, ln);
    stage_tile(out  + (size_t)m0 * D, s_g, ln);
    stage_wait();

    const v16bf* B = (const v16bf*)WgP;
    v8f acc[16];
#pragma unroll
    for (int t = 0; t < 16; ++t) acc[t] = (v8f){0,0,0,0,0,0,0,0};

    for (int ks = 0; ks < 16; ++ks) {            // K = 512: [prev ; agg]
        v16bf afrag = (ks < 8) ? make_afrag(s_p, ks, ln)
                               : make_afrag(s_g, ks - 8, ln);
        v16bf bcur = B[((size_t)(ks * 16)) * 32 + ln];
#pragma unroll
        for (int nt = 0; nt < 16; ++nt) {
            v16bf bnxt = bcur;
            if (nt < 15) bnxt = B[((size_t)(ks * 16 + nt + 1)) * 32 + ln];
            acc[nt] = __builtin_amdgcn_wmma_f32_16x16x32_bf16(
                false, afrag, false, bcur, (short)0, acc[nt], false, false);
            bcur = bnxt;
        }
    }
    int col = ln & 15, half = ln >> 4;
#pragma unroll
    for (int nt = 0; nt < 16; ++nt) {
        int n = nt * 16 + col;
        float b = bg[n];
#pragma unroll
        for (int r = 0; r < 8; ++r) {
            int rl = half * 8 + r;
            float p = s_p[rl * LPAD + n];
            float a = s_g[rl * LPAD + n];
            float z = 1.0f / (1.0f + __expf(-(acc[nt][r] + b)));
            out[(size_t)(m0 + rl) * D + n] = z * p + (1.0f - z) * a;
        }
    }
}

// ---------------------------------------------------------------------------
// launch
// ---------------------------------------------------------------------------
extern "C" void kernel_launch(void* const* d_in, const int* in_sizes, int n_in,
                              void* d_out, int out_size, void* d_ws, size_t ws_size,
                              hipStream_t stream) {
    const float*         enc  = (const float*)d_in[0];          // [2048,256,256]
    const unsigned char* mask = (const unsigned char*)d_in[1];  // [2048,256] bool
    const int*           idx  = (const int*)d_in[2];            // [2048,256]
    const float*         prev = (const float*)d_in[3];          // [50000,256]
    const float*         Wq   = (const float*)d_in[4];          // [256,256]
    const float*         Wg   = (const float*)d_in[5];          // [512,256]
    const float*         bg   = (const float*)d_in[6];          // [256]
    float* out = (float*)d_out;                                 // [50000,256]

    // workspace layout (all chunks 32B-aligned by construction)
    float*    q      = (float*)d_ws;                    // NN*D f32   (51.2 MB)
    float*    scores = q + (size_t)NN * D;              // NE  f32    (2 MB)
    unsigned* segmax = (unsigned*)(scores + NE);        // NN  u32
    float*    denom  = (float*)(segmax + NN);           // NN  f32
    bf16_t*   WqP    = (bf16_t*)(denom + NN);           // 128 frags * 512 bf16
    bf16_t*   WgP    = WqP + (size_t)128 * 32 * 16;     // 256 frags * 512 bf16

    k_init  <<<NN,        256, 0, stream>>>(out, segmax, denom);
    k_pack  <<<16,        256, 0, stream>>>(Wq, 128, WqP);      // 8 ks * 16 nt
    k_pack  <<<32,        256, 0, stream>>>(Wg, 256, WgP);      // 16 ks * 16 nt
    k_qproj <<<NN / 16,    32, 0, stream>>>(prev, WqP, q);
    k_scores<<<NE / 8,    256, 0, stream>>>(enc, mask, idx, q, scores, segmax);
    k_expden<<<NE / 256,  256, 0, stream>>>(idx, segmax, scores, denom);
    k_agg   <<<NE / 8,    256, 0, stream>>>(enc, idx, scores, denom, out);
    k_gate  <<<NN / 16,    32, 0, stream>>>(prev, WgP, bg, out);
}
